// ShiftwiseConv_26877905338440
// MI455X (gfx1250) — matrix operand order repaired and checked by reading, not compiled, not run on hardware
//
#include <hip/hip_runtime.h>

typedef __attribute__((ext_vector_type(16))) __bf16 bf16x16;
typedef __attribute__((ext_vector_type(8)))  float  v8f;

#define NB   32
#define C    128
#define HH   56
#define WW_  56
#define HW   (HH*WW_)          // 3136
#define NTAP 72                // 8 shifts * 9 kernel taps
#define TILE_N 128
#define NTILE  25              // ceil(3136/128)

// bytes
#define XB_BYTES  ((size_t)NB*HW*C*2)          // 25,690,112  bf16 NHWC input
#define WF_ELEMS  ((size_t)NTAP*8*4*32*16)     // 1,179,648   swizzled weights
#define WF_BYTES  (WF_ELEMS*2)                 // 2,359,296

// SHIFTS = [(-1,-1),(0,-1),(1,-1),(-1,0),(1,0),(-1,1),(0,1),(1,1)], dx->H, dy->W
__device__ __constant__ int c_dx[8] = {-1, 0, 1, -1, 1, -1, 0, 1};
__device__ __constant__ int c_dy[8] = {-1,-1,-1,  0, 0,  1, 1, 1};

__device__ __forceinline__ unsigned short f2bf(float f) {
    unsigned int u = __float_as_uint(f);
    u += 0x7FFFu + ((u >> 16) & 1u);           // round-to-nearest-even
    return (unsigned short)(u >> 16);
}

// ---- prepass: x (f32 NCHW) -> bf16 NHWC --------------------------------
__global__ void k_convert_x(const float* __restrict__ x, unsigned short* __restrict__ xb) {
    int idx = blockIdx.x * 256 + threadIdx.x;          // over N*C*H*W
    if (idx >= NB * C * HW) return;
    int w = idx % WW_;
    int h = (idx / WW_) % HH;
    int c = (idx / HW) % C;
    int n = idx / (C * HW);
    xb[((size_t)n * HW + h * WW_ + w) * C + c] = f2bf(x[idx]);
}

// ---- prepass: W [8][128][128][3][3] f32 -> per-tap A-fragment layout ----
// dst[t][mt(8)][kt(4)][lane(32)][j(16)], 16-bit A 16x32, k = 16*(lane/16) + j
__global__ void k_swizzle_w(const float* __restrict__ Wt, unsigned short* __restrict__ wf) {
    int d = blockIdx.x * 256 + threadIdx.x;
    if (d >= (int)WF_ELEMS) return;
    int j    =  d        & 15;
    int lane = (d >> 4)  & 31;
    int kt   = (d >> 9)  & 3;
    int mt   = (d >> 11) & 7;
    int t    =  d >> 14;
    int m = mt * 16 + (lane & 15);
    int k_local = (lane < 16) ? j : j + 16;
    int c  = kt * 32 + k_local;
    int i  = t / 9;
    int r  = t % 9;
    int kh = r / 3;
    int kw = r % 3;
    wf[d] = f2bf(Wt[(((size_t)(i * C + m) * C + c) * 3 + kh) * 3 + kw]);
}

// ---- zero page for OOB (zero-padded) taps ------------------------------
__global__ void k_init(unsigned int* zp) {
    if (threadIdx.x < 64) zp[threadIdx.x] = 0u;        // 256 B of zeros
}

// tap source pointer: roll = circular, conv pad = zero (redirect to zpage)
__device__ __forceinline__ const char* tap_bptr(const char* xb_n, const char* zpage,
                                                bool vq, int h, int w,
                                                int kh, int kw, int dx, int dy) {
    int a = h + kh - 1, b = w + kw - 1;
    bool v = vq && ((unsigned)a < HH) && ((unsigned)b < WW_);
    int hh = a - dx; hh += (hh < 0) ? HH : 0; hh -= (hh >= HH) ? HH : 0;
    int ww = b - dy; ww += (ww < 0) ? WW_ : 0; ww -= (ww >= WW_) ? WW_ : 0;
    return v ? (xb_n + (size_t)(hh * WW_ + ww) * (C * 2)) : zpage;
}

// ---- main implicit-GEMM conv: 72 taps x K=128, WMMA bf16, SW-pipelined --
__launch_bounds__(256)
__global__ void k_conv(const unsigned short* __restrict__ xb,
                       const unsigned short* __restrict__ wf,
                       const char* __restrict__ zpage,
                       float* __restrict__ out) {
    const int tile = blockIdx.x % NTILE;
    const int n    = blockIdx.x / NTILE;
    const int wave = threadIdx.x >> 5;
    const int lane = threadIdx.x & 31;
    const int wm    = wave >> 2;        // 0..1 : M half (64 rows)
    const int wn    = wave & 3;         // 0..3 : N quarter (32 px)
    const int lhalf = lane >> 4;        // B/K half + C/M-row half
    const int l15   = lane & 15;

    // two output pixels per lane (the two N-fragments of this wave)
    const int q0 = tile * TILE_N + wn * 32 + l15;
    const int q1 = q0 + 16;
    const bool vq0 = q0 < HW;
    const bool vq1 = q1 < HW;
    const int h0 = q0 / WW_, w0 = q0 % WW_;
    const int h1 = q1 / WW_, w1 = q1 % WW_;

    const char* xb_n = (const char*)xb + (size_t)n * HW * C * 2;
    const char* wfp  = (const char*)wf;            // +32768 B per tap

    v8f acc[4][2];
    const v8f vzero = {0.f,0.f,0.f,0.f,0.f,0.f,0.f,0.f};
#pragma unroll
    for (int f = 0; f < 4; ++f) { acc[f][0] = vzero; acc[f][1] = vzero; }

    int si = 0, kh = 0, kw = 0;
    const char* bp0 = tap_bptr(xb_n, zpage, vq0, h0, w0, 0, 0, c_dx[0], c_dy[0]);
    const char* bp1 = tap_bptr(xb_n, zpage, vq1, h1, w1, 0, 0, c_dx[0], c_dy[0]);
    const char* ap  = wfp + wm * 16384 + lane * 32;

    // pipeline prologue: load step (t=0, kt=0) into buffer 0
    bf16x16 Abuf[2][4], Bbuf[2][2];
    Bbuf[0][0] = *(const bf16x16*)(bp0 + lhalf * 32);
    Bbuf[0][1] = *(const bf16x16*)(bp1 + lhalf * 32);
#pragma unroll
    for (int f = 0; f < 4; ++f)
        Abuf[0][f] = *(const bf16x16*)(ap + f * 4096);

#pragma unroll 1
    for (int t = 0; t < NTAP; ++t) {
        // next tap's counters / pointers (clamped so tap-72 never reads OOB)
        int nkw = kw + 1, nkh = kh, nsi = si;
        if (nkw == 3) { nkw = 0; if (++nkh == 3) { nkh = 0; ++nsi; } }
        const int   csi  = (nsi < 8) ? nsi : 7;
        const char* nwfp = (t < NTAP - 1) ? (wfp + 32768) : wfp;
        const char* nbp0 = tap_bptr(xb_n, zpage, vq0, h0, w0, nkh, nkw, c_dx[csi], c_dy[csi]);
        const char* nbp1 = tap_bptr(xb_n, zpage, vq1, h1, w1, nkh, nkw, c_dx[csi], c_dy[csi]);
        const char* nap  = nwfp + wm * 16384 + lane * 32;

        __builtin_prefetch((t < NTAP - 2) ? (nwfp + 32768) : nwfp, 0, 3);
        __builtin_amdgcn_sched_barrier(0);     // keep addr math above the pipeline

#pragma unroll
        for (int kt = 0; kt < 4; ++kt) {
            const int cur = kt & 1, nxt = cur ^ 1;
            // issue loads for the NEXT k-step (or next tap's first k-step)
            const char* lb0  = (kt < 3) ? bp0 : nbp0;
            const char* lb1  = (kt < 3) ? bp1 : nbp1;
            const char* la   = (kt < 3) ? (ap + (kt + 1) * 1024) : nap;
            const int   koff = (kt < 3) ? (kt + 1) * 64 : 0;
            Bbuf[nxt][0] = *(const bf16x16*)(lb0 + koff + lhalf * 32);
            Bbuf[nxt][1] = *(const bf16x16*)(lb1 + koff + lhalf * 32);
#pragma unroll
            for (int f = 0; f < 4; ++f)
                Abuf[nxt][f] = *(const bf16x16*)(la + f * 4096);

            __builtin_amdgcn_sched_barrier(0); // loads(s+1) must stay above wmma(s)

            // 8 WMMAs on the already-resident buffer (hides the loads above)
#pragma unroll
            for (int f = 0; f < 4; ++f) {
                acc[f][0] = __builtin_amdgcn_wmma_f32_16x16x32_bf16(
                    false, Abuf[cur][f], false, Bbuf[cur][0], (short)0, acc[f][0], false, false);
                acc[f][1] = __builtin_amdgcn_wmma_f32_16x16x32_bf16(
                    false, Abuf[cur][f], false, Bbuf[cur][1], (short)0, acc[f][1], false, false);
            }
            __builtin_amdgcn_sched_barrier(0); // wmma(s) must stay above loads(s+2)
        }
        bp0 = nbp0; bp1 = nbp1; ap = nap; wfp = nwfp;
        kw = nkw; kh = nkh; si = nsi;
    }

    // store: C-frag element r -> row M = f*16 + lhalf*8 + r, col = q
    const int cbase = wm * 64 + lhalf * 8;
#pragma unroll
    for (int f = 0; f < 4; ++f) {
#pragma unroll
        for (int g = 0; g < 2; ++g) {
            const int q = g ? q1 : q0;
            if (q < HW) {
                float* op = out + ((size_t)n * C + cbase + f * 16) * HW + q;
#pragma unroll
                for (int r = 0; r < 8; ++r)
                    op[(size_t)r * HW] = acc[f][g][r];
            }
        }
    }
}

// ---- BatchNorm: per-channel batch stats --------------------------------
__global__ void k_bn_reduce(const float* __restrict__ out,
                            const float* __restrict__ gamma,
                            const float* __restrict__ beta,
                            float* __restrict__ scale,
                            float* __restrict__ shift) {
    __shared__ float s1[256], s2[256];
    const int c = blockIdx.x;
    float s = 0.f, ss = 0.f;
    for (int idx = threadIdx.x; idx < NB * HW; idx += 256) {
        int n = idx / HW, p = idx % HW;
        float v = out[((size_t)n * C + c) * HW + p];
        s += v; ss += v * v;
    }
    s1[threadIdx.x] = s; s2[threadIdx.x] = ss;
    __syncthreads();
    for (int o = 128; o > 0; o >>= 1) {
        if (threadIdx.x < o) {
            s1[threadIdx.x] += s1[threadIdx.x + o];
            s2[threadIdx.x] += s2[threadIdx.x + o];
        }
        __syncthreads();
    }
    if (threadIdx.x == 0) {
        const float inv = 1.0f / (float)(NB * HW);
        float mean = s1[0] * inv;
        float var  = s2[0] * inv - mean * mean;       // biased variance
        float sc = rsqrtf(var + 1e-5f) * gamma[c];
        scale[c] = sc;
        shift[c] = beta[c] - mean * sc;
    }
}

__global__ void k_bn_apply(float* __restrict__ out,
                           const float* __restrict__ scale,
                           const float* __restrict__ shift) {
    int idx = blockIdx.x * 256 + threadIdx.x;
    if (idx >= NB * C * HW) return;
    int c = (idx / HW) % C;
    out[idx] = out[idx] * scale[c] + shift[c];
}

extern "C" void kernel_launch(void* const* d_in, const int* in_sizes, int n_in,
                              void* d_out, int out_size, void* d_ws, size_t ws_size,
                              hipStream_t stream) {
    const float* x     = (const float*)d_in[0];
    const float* Wt    = (const float*)d_in[1];
    const float* gamma = (const float*)d_in[2];
    const float* beta  = (const float*)d_in[3];
    float* out = (float*)d_out;

    char* ws = (char*)d_ws;
    unsigned short* xb = (unsigned short*)ws;
    unsigned short* wf = (unsigned short*)(ws + XB_BYTES);
    char*  zpage       = ws + XB_BYTES + WF_BYTES;          // 256 B zeros
    float* scale       = (float*)(ws + XB_BYTES + WF_BYTES + 256);
    float* shift       = scale + 128;

    const int total = NB * C * HW;                          // 12,845,056
    k_convert_x<<<(total + 255) / 256, 256, 0, stream>>>(x, xb);
    k_swizzle_w<<<((int)WF_ELEMS + 255) / 256, 256, 0, stream>>>(Wt, wf);
    k_init<<<1, 64, 0, stream>>>((unsigned int*)zpage);
    k_conv<<<NB * NTILE, 256, 0, stream>>>(xb, wf, zpage, out);
    k_bn_reduce<<<C, 256, 0, stream>>>(out, gamma, beta, scale, shift);
    k_bn_apply<<<(total + 255) / 256, 256, 0, stream>>>(out, scale, shift);
}
